// GCN_33646773796939
// MI455X (gfx1250) — matrix operand order repaired and checked by reading, not compiled, not run on hardware
//
#include <hip/hip_runtime.h>

typedef __attribute__((ext_vector_type(16))) _Float16 v16h;
typedef __attribute__((ext_vector_type(8)))  float    v8f;

#define BB   2
#define NN   2048
#define CC   128
#define HH   4
#define HDIM 32
#define KNN  10

union HF16 { v16h v; _Float16 e[16]; };

// ------------------------------------------------------------------
// kNN: one thread per (b,n); insertion sort of K+1 smallest sq-dists.
// ------------------------------------------------------------------
__global__ __launch_bounds__(256) void knn_kernel(const float* __restrict__ coords,
                                                  int* __restrict__ idx)
{
    int t = blockIdx.x * blockDim.x + threadIdx.x;
    if (t >= BB * NN) return;
    int b = t / NN, n = t % NN;
    const float* cb = coords + (size_t)b * 3 * NN;
    float px = cb[n], py = cb[NN + n], pz = cb[2 * NN + n];
    float bd[KNN + 1]; int bi[KNN + 1];
    for (int i = 0; i <= KNN; i++) { bd[i] = 3.4e38f; bi[i] = 0; }
    for (int m = 0; m < NN; m++) {
        float dx = cb[m] - px, dy = cb[NN + m] - py, dz = cb[2 * NN + m] - pz;
        float d = dx * dx + dy * dy + dz * dz;
        if (d < bd[KNN]) {
            int j = KNN;
            while (j > 0 && bd[j - 1] > d) { bd[j] = bd[j - 1]; bi[j] = bi[j - 1]; j--; }
            bd[j] = d; bi[j] = m;
        }
    }
    int* op = idx + ((size_t)b * NN + n) * KNN;
    for (int k2 = 0; k2 < KNN; k2++) op[k2] = bi[k2 + 1];
}

// ------------------------------------------------------------------
// Weight prep: out_f16[m,k] = W[m, colofs+k] (- W[m, subofs+k])
// ------------------------------------------------------------------
__global__ __launch_bounds__(256) void cvtw_kernel(const float* __restrict__ W,
    int ldw, int colofs, int subofs, _Float16* __restrict__ out, int M, int Kd)
{
    int i = blockIdx.x * 256 + threadIdx.x;
    if (i >= M * Kd) return;
    int m = i / Kd, k = i % Kd;
    float w = W[(size_t)m * ldw + colofs + k];
    if (subofs >= 0) w -= W[(size_t)m * ldw + subofs + k];
    out[i] = (_Float16)w;
}

// ------------------------------------------------------------------
// Activation prep: (B,Cx,N) f32 -> (B,N,Cx) f16, tiled transpose.
// ------------------------------------------------------------------
__global__ __launch_bounds__(256) void tcvt_kernel(const float* __restrict__ src,
    _Float16* __restrict__ dst, int Cx)
{
    __shared__ float tile[32][33];
    const int b = blockIdx.z;
    const int n0 = blockIdx.x * 32, c0 = blockIdx.y * 32;
    const int tx = threadIdx.x & 31, ty = threadIdx.x >> 5;   // 32x8
    for (int i = 0; i < 32; i += 8)
        tile[ty + i][tx] = src[((size_t)b * Cx + c0 + ty + i) * NN + n0 + tx];
    __syncthreads();
    for (int i = 0; i < 32; i += 8)
        dst[((size_t)b * NN + n0 + ty + i) * Cx + c0 + tx] = (_Float16)tile[tx][ty + i];
}

// ------------------------------------------------------------------
// GEMM: Y[b] = Wf16 @ Xt[b]^T + bias.  Wf16:(M,Kd) f16 row-major,
// Xt:(B,NN,Kd) f16 (transposed activations), Y:(B,M,NN) f32.
// Double-buffered async global->LDS DMA (ASYNCcnt partial waits)
// overlapped with v_wmma_f32_16x16x32_f16. 128 thr = 4 waves = 64x64 C.
// ------------------------------------------------------------------
__global__ __launch_bounds__(128) void gemm_wmma_kernel(
    const _Float16* __restrict__ Wf16, const float* __restrict__ bias,
    const _Float16* __restrict__ Xt, float* __restrict__ Y, int M, int Kd)
{
    __shared__ __align__(16) _Float16 As[2][64][40];   // (m,k), 80B row = 5x16B
    __shared__ __align__(16) _Float16 Bs[2][64][40];   // (n,k)
    const int b = blockIdx.z;
    const _Float16* Xb = Xt + (size_t)b * NN * Kd;
    float*           Yb = Y  + (size_t)b * M  * NN;
    const int n0 = blockIdx.x * 64, m0 = blockIdx.y * 64;
    const int t = threadIdx.x, wave = t >> 5, lane = t & 31;
    const int hi = lane >> 4, lo = lane & 15;
    const unsigned laA = (unsigned)(size_t)(&As[0][0][0]);
    const unsigned laB = (unsigned)(size_t)(&Bs[0][0][0]);
    const unsigned BUFB = 64u * 40u * 2u;   // 5120 B per buffer

    auto issue = [&](int k0, int buf) {
        // 64x32 f16 tile = 256 16B chunks, 2 per thread, per operand.
        for (int ii = 0; ii < 2; ii++) {
            int ch = ii * 128 + t, r = ch >> 2, c8 = (ch & 3) * 8;
            unsigned long long ga =
                (unsigned long long)(Wf16 + (size_t)(m0 + r) * Kd + k0 + c8);
            unsigned la = laA + (unsigned)buf * BUFB + (unsigned)(r * 40 + c8) * 2u;
            asm volatile("global_load_async_to_lds_b128 %0, %1, off"
                         :: "v"(la), "v"(ga) : "memory");
            unsigned long long gb =
                (unsigned long long)(Xb + (size_t)(n0 + r) * Kd + k0 + c8);
            unsigned lb = laB + (unsigned)buf * BUFB + (unsigned)(r * 40 + c8) * 2u;
            asm volatile("global_load_async_to_lds_b128 %0, %1, off"
                         :: "v"(lb), "v"(gb) : "memory");
        }
    };

    v8f acc[4] = {};
    const int nk = Kd >> 5;
    issue(0, 0);
    for (int kt = 0; kt < nk; kt++) {
        const int buf = kt & 1;
        if (kt + 1 < nk) {
            issue((kt + 1) * 32, buf ^ 1);      // prefetch next tile (4 async/wave)
            asm volatile("s_wait_asynccnt 0x4" ::: "memory");  // drain current only
        } else {
            asm volatile("s_wait_asynccnt 0x0" ::: "memory");
        }
        __syncthreads();
        HF16 a;
        { int m = wave * 16 + lo;
          for (int j = 0; j < 16; j++) a.e[j] = As[buf][m][hi * 16 + j]; }
        for (int g = 0; g < 4; g++) {
            HF16 bf; int col = g * 16 + lo, kb2 = hi * 16;
            for (int j = 0; j < 16; j++) bf.e[j] = Bs[buf][col][kb2 + j];
            acc[g] = __builtin_amdgcn_wmma_f32_16x16x32_f16(
                false, a.v, false, bf.v, (short)0, acc[g], false, false);
        }
        __syncthreads();   // protect buf from next iteration's prefetch
    }
    float bvals[8];
    if (bias) {
        for (int r = 0; r < 8; r++) bvals[r] = bias[m0 + wave * 16 + hi * 8 + r];
    } else {
        for (int r = 0; r < 8; r++) bvals[r] = 0.f;
    }
    for (int g = 0; g < 4; g++) {
        int n = n0 + g * 16 + lo;
        for (int r = 0; r < 8; r++) {
            int m = m0 + wave * 16 + hi * 8 + r;
            Yb[(size_t)m * NN + n] = acc[g][r] + bvals[r];
        }
    }
}

// ------------------------------------------------------------------
// Instance-norm stats over (n[,k]) for y = P[o,n] (+ Q[o,idx[n,k]]).
// ------------------------------------------------------------------
__global__ __launch_bounds__(256) void gf_stats_kernel(
    const float* __restrict__ P, const float* __restrict__ Q,
    const int* __restrict__ idx, int Kn, int M,
    float* __restrict__ meanb, float* __restrict__ rstdb)
{
    __shared__ float ssum[256], ssq[256];
    const int o = blockIdx.x, b = blockIdx.y, t = threadIdx.x;
    const float* Pp = P + ((size_t)b * M + o) * NN;
    const float* Qp = Q ? Q + ((size_t)b * M + o) * NN : nullptr;
    const int* ib = idx ? idx + (size_t)b * NN * KNN : nullptr;
    float sum = 0.f, sq = 0.f;
    for (int n = t; n < NN; n += 256) {
        float pv = Pp[n];
        if (Qp) {
            const int* ii = ib + (size_t)n * Kn;
            for (int k2 = 0; k2 < Kn; k2++) {
                float v = pv + Qp[ii[k2]];
                sum += v; sq += v * v;
            }
        } else { sum += pv; sq += pv * pv; }
    }
    ssum[t] = sum; ssq[t] = sq; __syncthreads();
    for (int s2 = 128; s2 > 0; s2 >>= 1) {
        if (t < s2) { ssum[t] += ssum[t + s2]; ssq[t] += ssq[t + s2]; }
        __syncthreads();
    }
    if (t == 0) {
        float cnt = (float)NN * (Qp ? (float)Kn : 1.f);
        float m = ssum[0] / cnt;
        float var = ssq[0] / cnt - m * m;
        meanb[b * M + o] = m;
        rstdb[b * M + o] = rsqrtf(var + 1e-5f);
    }
}

// ------------------------------------------------------------------
// out[b,o,n] = max_k lrelu((P[o,n]+Q[o,idx[n,k]] - mu) * rstd)
// ------------------------------------------------------------------
__global__ __launch_bounds__(256) void gf_final_kernel(
    const float* __restrict__ P, const float* __restrict__ Q,
    const int* __restrict__ idx, int Kn,
    const float* __restrict__ meanb, const float* __restrict__ rstdb,
    float slope, int M, float* __restrict__ out)
{
    const int n = blockIdx.x * 256 + threadIdx.x;
    const int o = blockIdx.y, b = blockIdx.z;
    float mu = meanb[b * M + o], rs = rstdb[b * M + o];
    float pv = P[((size_t)b * M + o) * NN + n];
    float best;
    if (Q) {
        const int* ii = idx + ((size_t)b * NN + n) * KNN;
        const float* Qp = Q + ((size_t)b * M + o) * NN;
        best = -3.4e38f;
        for (int k2 = 0; k2 < Kn; k2++) {
            float v = pv + Qp[ii[k2]];
            v = (v - mu) * rs;
            v = v >= 0.f ? v : slope * v;
            best = fmaxf(best, v);
        }
    } else {
        float v = (pv - mu) * rs;
        best = v >= 0.f ? v : slope * v;
    }
    out[((size_t)b * M + o) * NN + n] = best;
}

// ------------------------------------------------------------------
// Flash attention per (b, h, 64 q rows). HD=32, channel c = d*H+h.
// ------------------------------------------------------------------
__global__ __launch_bounds__(128) void attn_kernel(
    const float* __restrict__ Qg, const float* __restrict__ Kg,
    const float* __restrict__ Vg, float* __restrict__ Og)
{
    __shared__ __align__(16) _Float16 Qs[64][34];
    __shared__ __align__(16) _Float16 Ks[64][34];
    __shared__ __align__(16) _Float16 Vs[32][66];
    __shared__ __align__(16) _Float16 Ps[64][66];
    const int b = blockIdx.z, h = blockIdx.y, n0 = blockIdx.x * 64;
    const int t = threadIdx.x, wave = t >> 5, lane = t & 31;
    const int hi = lane >> 4, lo = lane & 15;
    const int nfix = t & 63, dbase = t >> 6;
    const float scale = 0.17677669529663689f;  // 1/sqrt(32)

    {
        float tmp[16];
        for (int i = 0; i < 16; i++)
            tmp[i] = Qg[((size_t)b * CC + (2 * i + dbase) * HH + h) * NN + n0 + nfix];
        for (int i = 0; i < 16; i++)
            Qs[nfix][2 * i + dbase] = (_Float16)(tmp[i] * scale);
    }
    __syncthreads();
    HF16 aq;
    { int m = wave * 16 + lo;
      for (int j = 0; j < 16; j++) aq.e[j] = Qs[m][hi * 16 + j]; }

    float mrow[8], lrow[8];
    for (int r = 0; r < 8; r++) { mrow[r] = -3.4e38f; lrow[r] = 0.f; }
    v8f o0 = {}, o1 = {};

    for (int m0 = 0; m0 < NN; m0 += 64) {
        {
            float tk[16], tv[16];
            for (int i = 0; i < 16; i++) {
                size_t base = ((size_t)b * CC + (2 * i + dbase) * HH + h) * NN + m0 + nfix;
                tk[i] = Kg[base];
                tv[i] = Vg[base];
            }
            for (int i = 0; i < 16; i++) {
                Ks[nfix][2 * i + dbase] = (_Float16)tk[i];
                Vs[2 * i + dbase][nfix] = (_Float16)tv[i];
            }
        }
        __syncthreads();
        v8f s[4]; v8f zc = {};
        for (int g = 0; g < 4; g++) {
            HF16 bk; int col = g * 16 + lo, kb2 = hi * 16;
            for (int j = 0; j < 16; j++) bk.e[j] = Ks[col][kb2 + j];
            s[g] = __builtin_amdgcn_wmma_f32_16x16x32_f16(
                false, aq.v, false, bk.v, (short)0, zc, false, false);
        }
        float mnew[8];
        for (int r = 0; r < 8; r++) {
            float v = fmaxf(fmaxf(s[0][r], s[1][r]), fmaxf(s[2][r], s[3][r]));
            for (int off = 1; off < 16; off <<= 1) v = fmaxf(v, __shfl_xor(v, off, 32));
            mnew[r] = fmaxf(mrow[r], v);
        }
        for (int r = 0; r < 8; r++) {
            float sc = __expf(mrow[r] - mnew[r]);
            lrow[r] *= sc; o0[r] *= sc; o1[r] *= sc; mrow[r] = mnew[r];
        }
        for (int g = 0; g < 4; g++)
            for (int r = 0; r < 8; r++)
                s[g][r] = __expf(s[g][r] - mnew[r]);
        for (int r = 0; r < 8; r++) {
            float sm = s[0][r] + s[1][r] + s[2][r] + s[3][r];
            for (int off = 1; off < 16; off <<= 1) sm += __shfl_xor(sm, off, 32);
            lrow[r] += sm;
        }
        for (int g = 0; g < 4; g++)
            for (int r = 0; r < 8; r++)
                Ps[wave * 16 + 8 * hi + r][g * 16 + lo] = (_Float16)s[g][r];
        for (int ks = 0; ks < 64; ks += 32) {
            HF16 ap;
            { int m = wave * 16 + lo;
              for (int j = 0; j < 16; j++) ap.e[j] = Ps[m][ks + hi * 16 + j]; }
            for (int gd = 0; gd < 2; gd++) {
                HF16 bv; int col = gd * 16 + lo, kb2 = hi * 16;
                for (int j = 0; j < 16; j++) bv.e[j] = Vs[col][ks + kb2 + j];
                if (gd == 0)
                    o0 = __builtin_amdgcn_wmma_f32_16x16x32_f16(false, ap.v, false, bv.v, (short)0, o0, false, false);
                else
                    o1 = __builtin_amdgcn_wmma_f32_16x16x32_f16(false, ap.v, false, bv.v, (short)0, o1, false, false);
            }
        }
        __syncthreads();
    }
    for (int gd = 0; gd < 2; gd++)
        for (int r = 0; r < 8; r++) {
            int dd = gd * 16 + lo;
            int row = n0 + wave * 16 + 8 * hi + r;
            float val = (gd == 0 ? o0[r] : o1[r]) / lrow[r];
            Og[((size_t)b * CC + dd * HH + h) * NN + row] = val;
        }
}

// ------------------------------------------------------------------
__global__ __launch_bounds__(256) void copych_kernel(const float* __restrict__ src,
    float* __restrict__ dst, int Cs, int Cd, int cofs)
{
    int n = blockIdx.x * 256 + threadIdx.x;
    int c = blockIdx.y, b = blockIdx.z;
    dst[((size_t)b * Cd + cofs + c) * NN + n] = src[((size_t)b * Cs + c) * NN + n];
}

__global__ __launch_bounds__(256) void axpy_kernel(float* __restrict__ y,
    const float* __restrict__ x, int n)
{
    int i = blockIdx.x * 256 + threadIdx.x;
    if (i < n) y[i] += x[i];
}

// ==================================================================
extern "C" void kernel_launch(void* const* d_in, const int* in_sizes, int n_in,
                              void* d_out, int out_size, void* d_ws, size_t ws_size,
                              hipStream_t stream)
{
    (void)in_sizes; (void)n_in; (void)out_size; (void)ws_size;
    const size_t DSZ = (size_t)BB * CC * NN;    // 524288 floats per desc

    float* fws = (float*)d_ws;
    size_t off = 0;
    auto alloc = [&](size_t n) { float* p = fws + off; off += n; return p; };
    float* d0   = alloc(DSZ);
    float* d1   = alloc(DSZ);
    float* Pb   = alloc(2 * DSZ);
    float* Qb   = alloc(2 * DSZ);
    float* x1b  = alloc(DSZ);
    float* x2b  = alloc(2 * DSZ);
    float* t5   = alloc(4 * DSZ);
    float* qb   = alloc(DSZ);
    float* kbuf = alloc(DSZ);
    float* vb   = alloc(DSZ);
    float* ao   = alloc(DSZ);
    float* mg   = alloc(DSZ);
    _Float16* xt = (_Float16*)alloc(2 * DSZ);         // (B,N,<=512) f16
    _Float16* wA = (_Float16*)alloc(256 * 512 / 2);   // f16 weight image
    float* meanb = alloc(1024);
    float* rstdb = alloc(1024);
    int* idx0 = (int*)alloc((size_t)BB * NN * KNN);
    int* idx1 = (int*)alloc((size_t)BB * NN * KNN);

    auto cvtw = [&](const float* W, int ldw, int colofs, int subofs, int M, int Kd) {
        cvtw_kernel<<<(M * Kd + 255) / 256, 256, 0, stream>>>(W, ldw, colofs, subofs, wA, M, Kd);
    };
    auto tcvt = [&](const float* src, int Cx) {
        tcvt_kernel<<<dim3(NN / 32, Cx / 32, BB), 256, 0, stream>>>(src, xt, Cx);
    };
    auto gemm = [&](const float* bias, float* Y, int M, int Kd) {
        gemm_wmma_kernel<<<dim3(NN / 64, M / 64, BB), 128, 0, stream>>>(wA, bias, xt, Y, M, Kd);
    };
    auto stats = [&](const float* P, const float* Q, const int* idx, int Kn, int M) {
        gf_stats_kernel<<<dim3(M, BB), 256, 0, stream>>>(P, Q, idx, Kn, M, meanb, rstdb);
    };
    auto finalk = [&](const float* P, const float* Q, const int* idx, int Kn,
                      float slope, int M, float* out) {
        gf_final_kernel<<<dim3(NN / 256, M, BB), 256, 0, stream>>>(
            P, Q, idx, Kn, meanb, rstdb, slope, M, out);
    };
    auto copych = [&](const float* src, float* dst, int Cs, int Cd, int cofs) {
        copych_kernel<<<dim3(NN / 256, Cs, BB), 256, 0, stream>>>(src, dst, Cs, Cd, cofs);
    };

    hipMemcpyAsync(d0, d_in[2], DSZ * sizeof(float), hipMemcpyDeviceToDevice, stream);
    hipMemcpyAsync(d1, d_in[3], DSZ * sizeof(float), hipMemcpyDeviceToDevice, stream);

    knn_kernel<<<(BB * NN + 255) / 256, 256, 0, stream>>>((const float*)d_in[0], idx0);
    knn_kernel<<<(BB * NN + 255) / 256, 256, 0, stream>>>((const float*)d_in[1], idx1);

    // self layer: factorized edge-conv  y = (Wa-Wb)x(:,n) + Wb x(:,idx)
    auto self_layer = [&](const float* W1, const float* W2, const float* W3,
                          float* d, const int* idx) {
        tcvt(d, CC);
        cvtw(W1, 2 * CC, 0, CC, CC, CC);   gemm(nullptr, Pb, CC, CC);
        cvtw(W1, 2 * CC, CC, -1, CC, CC);  gemm(nullptr, Qb, CC, CC);
        stats(Pb, Qb, idx, KNN, CC);
        finalk(Pb, Qb, idx, KNN, 0.2f, CC, x1b);
        tcvt(x1b, CC);
        cvtw(W2, 2 * CC, 0, CC, 2 * CC, CC);   gemm(nullptr, Pb, 2 * CC, CC);
        cvtw(W2, 2 * CC, CC, -1, 2 * CC, CC);  gemm(nullptr, Qb, 2 * CC, CC);
        stats(Pb, Qb, idx, KNN, 2 * CC);
        finalk(Pb, Qb, idx, KNN, 0.2f, 2 * CC, x2b);
        copych(d,   t5, CC,     4 * CC, 0);
        copych(x1b, t5, CC,     4 * CC, CC);
        copych(x2b, t5, 2 * CC, 4 * CC, 2 * CC);
        tcvt(t5, 4 * CC);
        cvtw(W3, 4 * CC, 0, -1, CC, 4 * CC);   gemm(nullptr, Pb, CC, 4 * CC);
        stats(Pb, nullptr, nullptr, 1, CC);
        finalk(Pb, nullptr, nullptr, 1, 0.2f, CC, d);
    };

    auto cross_layer = [&](int pi, float* dA, const float* dB) {
        const float* pW0 = (const float*)d_in[pi + 0];
        const float* pb0 = (const float*)d_in[pi + 1];
        const float* pW1 = (const float*)d_in[pi + 2];
        const float* pb1 = (const float*)d_in[pi + 3];
        const float* pW2 = (const float*)d_in[pi + 4];
        const float* pb2 = (const float*)d_in[pi + 5];
        const float* mW  = (const float*)d_in[pi + 6];
        const float* mb  = (const float*)d_in[pi + 7];
        const float* cW1 = (const float*)d_in[pi + 8];
        const float* cb1 = (const float*)d_in[pi + 9];
        const float* cW2 = (const float*)d_in[pi + 10];
        const float* cb2 = (const float*)d_in[pi + 11];
        tcvt(dA, CC);
        cvtw(pW0, CC, 0, -1, CC, CC);  gemm(pb0, qb, CC, CC);
        tcvt(dB, CC);
        cvtw(pW1, CC, 0, -1, CC, CC);  gemm(pb1, kbuf, CC, CC);
        cvtw(pW2, CC, 0, -1, CC, CC);  gemm(pb2, vb, CC, CC);
        attn_kernel<<<dim3(NN / 64, HH, BB), 128, 0, stream>>>(qb, kbuf, vb, ao);
        tcvt(ao, CC);
        cvtw(mW, CC, 0, -1, CC, CC);   gemm(mb, mg, CC, CC);
        copych(dA, t5, CC, 2 * CC, 0);
        copych(mg, t5, CC, 2 * CC, CC);
        tcvt(t5, 2 * CC);
        cvtw(cW1, 2 * CC, 0, -1, 2 * CC, 2 * CC);  gemm(cb1, Pb, 2 * CC, 2 * CC);
        stats(Pb, nullptr, nullptr, 1, 2 * CC);
        finalk(Pb, nullptr, nullptr, 1, 0.0f, 2 * CC, Qb);
        tcvt(Qb, 2 * CC);
        cvtw(cW2, 2 * CC, 0, -1, CC, 2 * CC);      gemm(cb2, ao, CC, 2 * CC);
        axpy_kernel<<<(int)((DSZ + 255) / 256), 256, 0, stream>>>(dA, ao, (int)DSZ);
    };

    self_layer((const float*)d_in[4], (const float*)d_in[5], (const float*)d_in[6], d0, idx0);
    self_layer((const float*)d_in[4], (const float*)d_in[5], (const float*)d_in[6], d1, idx1);
    cross_layer(7, d0, d1);
    cross_layer(7, d1, d0);
    self_layer((const float*)d_in[19], (const float*)d_in[20], (const float*)d_in[21], d0, idx0);
    self_layer((const float*)d_in[19], (const float*)d_in[20], (const float*)d_in[21], d1, idx1);
    cross_layer(22, d0, d1);
    cross_layer(22, d1, d0);

    float* out = (float*)d_out;
    hipMemcpyAsync(out,       d0, DSZ * sizeof(float), hipMemcpyDeviceToDevice, stream);
    hipMemcpyAsync(out + DSZ, d1, DSZ * sizeof(float), hipMemcpyDeviceToDevice, stream);
}
